// DeepRoute_32306744000792
// MI455X (gfx1250) — compile-verified, matrix operand order checked by reading, not run on hardware
//
#include <hip/hip_runtime.h>
#include <hip/hip_bf16.h>
#include <math.h>

// Problem constants (from reference)
#define Bsz 512
#define Esz 256
#define Hsz 256
#define Lsz 32
#define H4  1024
#define KX  512   // E + H

typedef __attribute__((ext_vector_type(16))) __bf16 v16bf;
typedef __attribute__((ext_vector_type(8)))  float  v8f;

struct alignas(16) U128 { unsigned int w[4]; };
union BFfrag { v16bf v; U128 q[2]; };

__device__ __forceinline__ unsigned short f2bf(float f) {
    unsigned int u = __float_as_uint(f);
    u += 0x7FFFu + ((u >> 16) & 1u);          // round-to-nearest-even
    return (unsigned short)(u >> 16);
}
__device__ __forceinline__ float sigf(float x) { return 1.0f / (1.0f + __expf(-x)); }

// ---------------- setup kernels ----------------

__global__ void k_f32_to_bf16(const float* __restrict__ src, unsigned short* __restrict__ dst, int n) {
    int t = blockIdx.x * 256 + threadIdx.x;
    if (t < n) dst[t] = f2bf(src[t]);
}

__global__ void k_pack_wcat(const float* __restrict__ wih, const float* __restrict__ whh,
                            const float* __restrict__ bih, const float* __restrict__ bhh,
                            unsigned short* __restrict__ wcat, float* __restrict__ bcat) {
    int t = blockIdx.x * 256 + threadIdx.x;           // H4*KX threads
    if (t < H4 * KX) {
        int n = t >> 9, k = t & 511;
        float v = (k < Esz) ? wih[n * Esz + k] : whh[n * Hsz + (k - Esz)];
        wcat[t] = f2bf(v);
    }
    if (t < H4) bcat[t] = bih[t] + bhh[t];
}

__global__ void k_init_xh(const float* __restrict__ x0, const float* __restrict__ h0,
                          unsigned short* __restrict__ xh) {
    int t = blockIdx.x * 256 + threadIdx.x;           // Bsz*KX threads
    int b = t >> 9, c = t & 511;
    float v = (c < Esz) ? x0[b * Esz + c] : h0[b * Hsz + (c - Esz)];
    xh[t] = f2bf(v);
}

__global__ void k_init_mask(const unsigned char* __restrict__ vm, int* __restrict__ mask) {
    int t = blockIdx.x * 256 + threadIdx.x;           // Bsz*Lsz threads, 1 wave = 1 row
    int l = t & 31;
    int m = vm[t] ? 1 : 0;
    unsigned long long bal = __ballot(m != 0);
    if (((bal & 0xffffffffull) == 0xffffffffull) && l == 31) m = 0;  // _mask_modify
    mask[t] = m;
}

// ---------------- WMMA GEMM:  C[M,N] = A[M,K](bf16,lda) * W[N,K]^T(bf16) + bias ----------------
// 256 threads = 8 waves arranged 2(M)x4(N); each wave computes a 32x64 macro-tile
// (2x4 register blocking: 8 f32 accumulators, A-frags reused 4x, B-frags reused 2x)
// via v_wmma_f32_16x16x32_bf16. Block tile = 64(M) x 256(N).

__global__ __launch_bounds__(256)
void k_gemm_bf16(const unsigned short* __restrict__ A, int lda,
                 const unsigned short* __restrict__ W,
                 const float* __restrict__ bias,
                 float* __restrict__ C, int N, int K) {
    const int wave = threadIdx.x >> 5;
    const int lane = threadIdx.x & 31;
    const int mg = wave >> 2;                         // 0..1
    const int ng = wave & 3;                          // 0..3
    const int m0 = blockIdx.x * 64 + mg * 32;
    const int n0 = blockIdx.y * 256 + ng * 64;
    const int row  = lane & 15;
    const int ksub = (lane >> 4) << 3;                // 0 | 8 (lanes 0-15 vs 16-31)
    const unsigned short* ap0 = A + (size_t)(m0 + row) * lda + ksub;
    const unsigned short* ap1 = ap0 + (size_t)16 * lda;
    const unsigned short* wp0 = W + (size_t)(n0 + row) * K + ksub;
    v8f acc[2][4] = {};
    for (int k0 = 0; k0 < K; k0 += 32) {
        BFfrag a0, a1;
        a0.q[0] = *(const U128*)(ap0 + k0);           // K = k0+ksub .. +7
        a0.q[1] = *(const U128*)(ap0 + k0 + 16);      // K = k0+ksub+16 .. +23
        a1.q[0] = *(const U128*)(ap1 + k0);
        a1.q[1] = *(const U128*)(ap1 + k0 + 16);
#pragma unroll
        for (int j = 0; j < 4; ++j) {
            BFfrag b;
            const unsigned short* wp = wp0 + (size_t)(16 * j) * K + k0;
            b.q[0] = *(const U128*)(wp);
            b.q[1] = *(const U128*)(wp + 16);
            acc[0][j] = __builtin_amdgcn_wmma_f32_16x16x32_bf16(false, a0.v, false, b.v,
                                                                (short)0, acc[0][j], false, false);
            acc[1][j] = __builtin_amdgcn_wmma_f32_16x16x32_bf16(false, a1.v, false, b.v,
                                                                (short)0, acc[1][j], false, false);
        }
    }
    const int nn = lane & 15;
    const int mb = (lane >> 4) << 3;
#pragma unroll
    for (int i = 0; i < 2; ++i) {
#pragma unroll
        for (int j = 0; j < 4; ++j) {
            const int n = n0 + 16 * j + nn;
            const float bv = bias ? bias[n] : 0.0f;
            float* cp = C + (size_t)(m0 + 16 * i + mb) * N + n;
#pragma unroll
            for (int r = 0; r < 8; ++r)
                cp[(size_t)r * N] = acc[i][j][r] + bv;
        }
    }
}

// ---------------- per-step elementwise kernels ----------------

__global__ void k_lstm_update(const float* __restrict__ gates, const float* __restrict__ cin,
                              float* __restrict__ cout, unsigned short* __restrict__ xh) {
    int t = blockIdx.x * 256 + threadIdx.x;           // Bsz*Hsz threads
    int b = t >> 8, h = t & 255;
    const float* g = gates + b * H4;
    float ig = sigf(g[h]);
    float fg = sigf(g[h + Hsz]);
    float gg = tanhf(g[h + 2 * Hsz]);
    float og = sigf(g[h + 3 * Hsz]);
    float c  = fg * cin[t] + ig * gg;
    cout[t] = c;
    xh[b * KX + Esz + h] = f2bf(og * tanhf(c));       // h_new feeds next LSTM + Wq GEMM
}

// u[b,l] = sum_h v[h]*tanh(qq[b,h]+e[l,b,h]); one wave per (b,l)
__global__ void k_attn_scores(const float* __restrict__ qq, const float* __restrict__ e,
                              const float* __restrict__ v, const int* __restrict__ mask,
                              float* __restrict__ logits, int use_tanh) {
    int wid  = blockIdx.x * 8 + (threadIdx.x >> 5);   // Bsz*Lsz waves
    int lane = threadIdx.x & 31;
    int b = wid >> 5, l = wid & 31;
    const float* qrow = qq + b * Hsz;
    const float* erow = e + (l * Bsz + b) * Hsz;
    float s = 0.0f;
#pragma unroll
    for (int i = 0; i < 8; ++i) {
        int h = lane + i * 32;
        s += v[h] * tanhf(qrow[h] + erow[h]);
    }
    for (int off = 16; off > 0; off >>= 1) s += __shfl_down(s, off, 32);
    if (lane == 0) {
        float u = use_tanh ? 10.0f * tanhf(s) : s;
        if (mask[b * 32 + l]) u = -__builtin_inff();
        logits[b * 32 + l] = u;
    }
}

// softmax over L then g_l[b,h] = sum_l p[l]*e_gl[l,b,h]; one block per b
__global__ void k_glimpse_readout(const float* __restrict__ logits, const float* __restrict__ e_gl,
                                  unsigned short* __restrict__ glbf) {
    int b = blockIdx.x, tid = threadIdx.x;
    __shared__ float p[Lsz];
    if (tid < 32) {
        float x = logits[b * 32 + tid];
        float m = x;
        for (int off = 16; off > 0; off >>= 1) m = fmaxf(m, __shfl_xor(m, off, 32));
        float ex = __expf(x - m);                     // exp(-inf)=0 for masked
        float sm = ex;
        for (int off = 16; off > 0; off >>= 1) sm += __shfl_xor(sm, off, 32);
        p[tid] = ex / sm;
    }
    __syncthreads();
    float g = 0.0f;
    const float* eb = e_gl + b * Hsz + tid;
#pragma unroll 4
    for (int l = 0; l < Lsz; ++l) g += p[l] * eb[l * Bsz * Hsz];
    glbf[b * Hsz + tid] = f2bf(g);
}

// log_softmax -> out, argmax -> sels, mask update (+_mask_modify), gather next x; one block per b
__global__ void k_finalize(const float* __restrict__ logits, int* __restrict__ mask,
                           const unsigned short* __restrict__ embb, unsigned short* __restrict__ xh,
                           float* __restrict__ out_logp, int* __restrict__ out_sels, int step) {
    int b = blockIdx.x, tid = threadIdx.x;
    __shared__ int s_idx;
    if (tid < 32) {
        int l = tid;
        float x = logits[b * 32 + l];
        float m = x;
        for (int off = 16; off > 0; off >>= 1) m = fmaxf(m, __shfl_xor(m, off, 32));
        float ex = __expf(x - m);
        float sm = ex;
        for (int off = 16; off > 0; off >>= 1) sm += __shfl_xor(sm, off, 32);
        out_logp[(b * Lsz + step) * Lsz + l] = x - m - __logf(sm);
        // argmax, lowest-index tie break
        float bv = x; int bi = l;
        for (int off = 16; off > 0; off >>= 1) {
            float ov = __shfl_xor(bv, off, 32);
            int   oi = __shfl_xor(bi, off, 32);
            if (ov > bv || (ov == bv && oi < bi)) { bv = ov; bi = oi; }
        }
        int mk = mask[b * 32 + l] | (l == bi);
        unsigned long long bal = __ballot(mk != 0);
        if (((bal & 0xffffffffull) == 0xffffffffull) && l == 31) mk = 0;  // _mask_modify
        mask[b * 32 + l] = mk;
        if (l == 0) { s_idx = bi; out_sels[b * Lsz + step] = bi; }
    }
    __syncthreads();
    xh[b * KX + tid] = embb[(s_idx * Bsz + b) * Esz + tid];  // next decoder input (bf16)
}

// ---------------- host ----------------

extern "C" void kernel_launch(void* const* d_in, const int* in_sizes, int n_in,
                              void* d_out, int out_size, void* d_ws, size_t ws_size,
                              hipStream_t stream) {
    const float* x0   = (const float*)d_in[0];
    const float* emb  = (const float*)d_in[1];
    const float* h0   = (const float*)d_in[2];
    const float* c0   = (const float*)d_in[3];
    const float* ctx  = (const float*)d_in[4];
    const unsigned char* vrm = (const unsigned char*)d_in[5];
    const float* Wih  = (const float*)d_in[6];
    const float* Whh  = (const float*)d_in[7];
    const float* bih  = (const float*)d_in[8];
    const float* bhh  = (const float*)d_in[9];
    const float* pWq  = (const float*)d_in[10];
    const float* pbq  = (const float*)d_in[11];
    const float* pWr  = (const float*)d_in[12];
    const float* pbr  = (const float*)d_in[13];
    const float* pv   = (const float*)d_in[14];
    const float* gWq  = (const float*)d_in[15];
    const float* gbq  = (const float*)d_in[16];
    const float* gWr  = (const float*)d_in[17];
    const float* gbr  = (const float*)d_in[18];
    const float* gv   = (const float*)d_in[19];

    char* w = (char*)d_ws;
    size_t off = 0;
    auto alloc = [&](size_t bytes) { char* p = w + off; off += (bytes + 255) & ~size_t(255); return p; };
    unsigned short* wcat  = (unsigned short*)alloc((size_t)H4 * KX * 2);
    unsigned short* gWqb  = (unsigned short*)alloc((size_t)Hsz * Hsz * 2);
    unsigned short* pWqb  = (unsigned short*)alloc((size_t)Hsz * Hsz * 2);
    unsigned short* gWrb  = (unsigned short*)alloc((size_t)Hsz * Hsz * 2);
    unsigned short* pWrb  = (unsigned short*)alloc((size_t)Hsz * Hsz * 2);
    unsigned short* ctxb  = (unsigned short*)alloc((size_t)Lsz * Bsz * Hsz * 2);
    unsigned short* embb  = (unsigned short*)alloc((size_t)Lsz * Bsz * Esz * 2);
    float*          e_gl  = (float*)alloc((size_t)Lsz * Bsz * Hsz * 4);
    float*          e_pt  = (float*)alloc((size_t)Lsz * Bsz * Hsz * 4);
    unsigned short* xh    = (unsigned short*)alloc((size_t)Bsz * KX * 2);
    float*          gates = (float*)alloc((size_t)Bsz * H4 * 4);
    float*          cst   = (float*)alloc((size_t)Bsz * Hsz * 4);
    float*          qq    = (float*)alloc((size_t)Bsz * Hsz * 4);
    float*          glog  = (float*)alloc((size_t)Bsz * Lsz * 4);
    unsigned short* glbf  = (unsigned short*)alloc((size_t)Bsz * Hsz * 2);
    float*          plog  = (float*)alloc((size_t)Bsz * Lsz * 4);
    int*            mask  = (int*)alloc((size_t)Bsz * Lsz * 4);
    float*          bcat  = (float*)alloc((size_t)H4 * 4);

    float* out_logp = (float*)d_out;                       // [B, L, L]
    int*   out_sels = (int*)(out_logp + (size_t)Bsz * Lsz * Lsz);

    const int TB = 256;
    auto grid1 = [](int n) { return dim3((n + 255) / 256); };

    // one-time conversions / packing
    k_f32_to_bf16<<<grid1(Lsz * Bsz * Hsz), TB, 0, stream>>>(ctx, ctxb, Lsz * Bsz * Hsz);
    k_f32_to_bf16<<<grid1(Lsz * Bsz * Esz), TB, 0, stream>>>(emb, embb, Lsz * Bsz * Esz);
    k_f32_to_bf16<<<grid1(Hsz * Hsz), TB, 0, stream>>>(gWq, gWqb, Hsz * Hsz);
    k_f32_to_bf16<<<grid1(Hsz * Hsz), TB, 0, stream>>>(pWq, pWqb, Hsz * Hsz);
    k_f32_to_bf16<<<grid1(Hsz * Hsz), TB, 0, stream>>>(gWr, gWrb, Hsz * Hsz);
    k_f32_to_bf16<<<grid1(Hsz * Hsz), TB, 0, stream>>>(pWr, pWrb, Hsz * Hsz);
    k_pack_wcat<<<grid1(H4 * KX), TB, 0, stream>>>(Wih, Whh, bih, bhh, wcat, bcat);
    k_init_xh<<<grid1(Bsz * KX), TB, 0, stream>>>(x0, h0, xh);
    k_init_mask<<<grid1(Bsz * Lsz), TB, 0, stream>>>(vrm, mask);

    // hoisted context projections (step-invariant): e = ctx @ Wr^T + br, both heads
    k_gemm_bf16<<<dim3(Lsz * Bsz / 64, Hsz / 256), TB, 0, stream>>>(ctxb, Hsz, gWrb, gbr, e_gl, Hsz, Hsz);
    k_gemm_bf16<<<dim3(Lsz * Bsz / 64, Hsz / 256), TB, 0, stream>>>(ctxb, Hsz, pWrb, pbr, e_pt, Hsz, Hsz);

    for (int step = 0; step < Lsz; ++step) {
        // fused LSTM GEMM: gates = [x|h] @ [Wih|Whh]^T + (bih+bhh)
        k_gemm_bf16<<<dim3(Bsz / 64, H4 / 256), TB, 0, stream>>>(xh, KX, wcat, bcat, gates, H4, KX);
        k_lstm_update<<<grid1(Bsz * Hsz), TB, 0, stream>>>(gates, step == 0 ? c0 : cst, cst, xh);
        // glimpse
        k_gemm_bf16<<<dim3(Bsz / 64, Hsz / 256), TB, 0, stream>>>(xh + Esz, KX, gWqb, gbq, qq, Hsz, Hsz);
        k_attn_scores<<<dim3(Bsz * Lsz / 8), TB, 0, stream>>>(qq, e_gl, gv, mask, glog, 0);
        k_glimpse_readout<<<dim3(Bsz), TB, 0, stream>>>(glog, e_gl, glbf);
        // pointer
        k_gemm_bf16<<<dim3(Bsz / 64, Hsz / 256), TB, 0, stream>>>(glbf, Hsz, pWqb, pbq, qq, Hsz, Hsz);
        k_attn_scores<<<dim3(Bsz * Lsz / 8), TB, 0, stream>>>(qq, e_pt, pv, mask, plog, 1);
        // log_softmax + argmax + mask update + next-input gather
        k_finalize<<<dim3(Bsz), TB, 0, stream>>>(plog, mask, embb, xh, out_logp, out_sels, step);
    }
    (void)in_sizes; (void)n_in; (void)out_size; (void)ws_size;
}
